// Block_81415400063592
// MI455X (gfx1250) — matrix operand order repaired and checked by reading, not compiled, not run on hardware
//
#include <hip/hip_runtime.h>

#define DIMSZ   1024
#define NHEADS  16
#define DPH     64
#define BATCH   4
#define SEQ     2048

typedef _Float16 v16h __attribute__((ext_vector_type(16)));
typedef _Float16 v8h  __attribute__((ext_vector_type(8)));
typedef float    v8f  __attribute__((ext_vector_type(8)));

static __device__ __forceinline__ v8f wmma_f16(const v16h& a, const v16h& b, const v8f& c) {
    return __builtin_amdgcn_wmma_f32_16x16x32_f16(false, a, false, b, (short)0, c, false, false);
}

// Async copy of 16 bytes global -> LDS (ASYNCcnt tracked, no VGPR round-trip).
static __device__ __forceinline__ void async_copy_b128(const void* gptr, const void* lptr) {
    unsigned loff = (unsigned)(unsigned long long)lptr;   // LDS offset = low 32 bits
    asm volatile("global_load_async_to_lds_b128 %0, %1, off"
                 :: "v"(loff), "v"(gptr) : "memory");
}
static __device__ __forceinline__ void wait_async0() {
    asm volatile("s_wait_asynccnt 0x0" ::: "memory");
}

// ---------------------------------------------------------------------------
// Generic WMMA GEMM:  C[M,N] = act(A[M,K] @ W[K,N] + bias[N])
// Block: 256 threads (8 waves). Tile 128x128, K-step 32.
// Wave (wm in 0..3, wn in 0..1) computes rows [wm*32,+32) x cols [wn*64,+64).
// A tile staged fp32 via async-to-LDS; W tile staged transposed f16 manually.
// ---------------------------------------------------------------------------
#define ASTRF 36  // fp32 A-tile row stride (32 + 4 pad); row*144B is 16B aligned
#define GSTR  40  // f16 W-tile row stride (32 + 8 pad); row*80B is 16B aligned

__global__ __launch_bounds__(256) void gemm_wmma_kernel(
    const float* __restrict__ A, const float* __restrict__ W,
    const float* __restrict__ bias, float* __restrict__ C,
    int M, int N, int K, int relu)
{
    __shared__ float    lAf[128 * ASTRF];
    __shared__ _Float16 lB[128 * GSTR];   // transposed W tile: [n][k]

    const int tid  = threadIdx.x;
    const int wave = tid >> 5;
    const int lane = tid & 31;
    const int wm   = wave >> 1;
    const int wn   = wave & 1;
    const int brow = blockIdx.y * 128;
    const int bcol = blockIdx.x * 128;

    v8f acc[2][4];
    for (int tm = 0; tm < 2; ++tm)
        for (int tn = 0; tn < 4; ++tn)
            for (int j = 0; j < 8; ++j) acc[tm][tn][j] = 0.0f;

    const int l15   = lane & 15;
    const int aksel = (lane < 16) ? 0 : 8;    // A-frag K select
    const int bksel = (lane < 16) ? 0 : 8;    // B-frag K select

    for (int k0 = 0; k0 < K; k0 += 32) {
        __syncthreads();
        // Stage A tile (128 x 32 fp32) via async DMA to LDS
        for (int it = 0; it < 4; ++it) {
            int i  = tid + it * 256;          // 1024 b128 chunks
            int r  = i >> 3;
            int c4 = (i & 7) * 4;
            async_copy_b128(A + (size_t)(brow + r) * K + k0 + c4,
                            &lAf[r * ASTRF + c4]);
        }
        // Stage W tile transposed (32 x 128 f32 -> f16 [n][k])
        for (int it = 0; it < 4; ++it) {
            int i  = tid + it * 256;
            int kr = i >> 5;
            int c4 = (i & 31) * 4;
            float4 f = *(const float4*)(W + (size_t)(k0 + kr) * N + bcol + c4);
            lB[(c4 + 0) * GSTR + kr] = (_Float16)f.x;
            lB[(c4 + 1) * GSTR + kr] = (_Float16)f.y;
            lB[(c4 + 2) * GSTR + kr] = (_Float16)f.z;
            lB[(c4 + 3) * GSTR + kr] = (_Float16)f.w;
        }
        wait_async0();
        __syncthreads();

        v16h af[2], bf[4];
        for (int tm = 0; tm < 2; ++tm) {
            const float* p = &lAf[(wm * 32 + tm * 16 + l15) * ASTRF + aksel];
            for (int e = 0; e < 8; ++e) {
                af[tm][e]     = (_Float16)p[e];
                af[tm][8 + e] = (_Float16)p[16 + e];
            }
        }
        for (int tn = 0; tn < 4; ++tn) {
            const _Float16* p = &lB[(wn * 64 + tn * 16 + l15) * GSTR + bksel];
            v8h lo = *(const v8h*)p;
            v8h hi = *(const v8h*)(p + 16);
            for (int e = 0; e < 8; ++e) { bf[tn][e] = lo[e]; bf[tn][8 + e] = hi[e]; }
        }
        for (int tm = 0; tm < 2; ++tm)
            for (int tn = 0; tn < 4; ++tn)
                acc[tm][tn] = wmma_f16(af[tm], bf[tn], acc[tm][tn]);
    }

    const int moff = (lane < 16) ? 0 : 8;
    for (int tm = 0; tm < 2; ++tm) {
        for (int tn = 0; tn < 4; ++tn) {
            int col  = bcol + wn * 64 + tn * 16 + l15;
            float bv = bias ? bias[col] : 0.0f;
            for (int j = 0; j < 8; ++j) {
                int row = brow + wm * 32 + tm * 16 + moff + j;
                float v = acc[tm][tn][j] + bv;
                if (relu) v = fmaxf(v, 0.0f);
                C[(size_t)row * N + col] = v;
            }
        }
    }
}

// ---------------------------------------------------------------------------
// Flash-style causal attention. Grid: (SEQ/128, NHEADS, BATCH), 256 threads.
// Each wave owns 16 q-rows. Q pre-scaled by 1/DPH (reference's double-scale).
// K block staged fp32 via async-to-LDS (shared by all 8 waves);
// V block staged transposed f16; P re-laid out via per-wave LDS.
// ---------------------------------------------------------------------------
#define KSTRF 68  // fp32 K-tile row stride (64 + 4); row*272B is 16B aligned
#define VSTR  80  // f16 Vt row stride; row*160B is 16B aligned
#define PSTR  80  // f16 per-wave P tile row stride

__global__ __launch_bounds__(256) void attn_kernel(
    const float* __restrict__ Q, const float* __restrict__ Kmat,
    const float* __restrict__ V, float* __restrict__ O)
{
    __shared__ float    lK[64 * KSTRF];         // fp32 K block: [k_local][d]
    __shared__ _Float16 lVt[DPH * VSTR];        // transposed V block: [d][k_local]
    __shared__ _Float16 lP[8 * 16 * PSTR];      // per-wave 16x64 prob tile

    const int tid  = threadIdx.x;
    const int wave = tid >> 5;
    const int lane = tid & 31;
    const int b    = blockIdx.z;
    const int h    = blockIdx.y;
    const int q0   = blockIdx.x * 128;
    const int qw   = q0 + wave * 16;

    const float* qbase = Q + (size_t)b * SEQ * DIMSZ + h * DPH;
    const float* kbase = Kmat + (size_t)b * SEQ * DIMSZ + h * DPH;
    const float* vbase = V + (size_t)b * SEQ * DIMSZ + h * DPH;
    float*       obase = O + (size_t)b * SEQ * DIMSZ + h * DPH;

    const int l15   = lane & 15;
    const int aksel = (lane < 16) ? 0 : 8;
    const int bksel = (lane < 16) ? 0 : 8;
    const int moff  = (lane < 16) ? 0 : 8;

    // Preload Q fragments (A-layout), folding in the 1/DPH score scale.
    v16h aq[2];
    {
        const float* qp = qbase + (size_t)(qw + l15) * DIMSZ;
        const float sc = 1.0f / (float)DPH;
        for (int s = 0; s < 2; ++s)
            for (int e = 0; e < 8; ++e) {
                aq[s][e]     = (_Float16)(qp[s * 32 + aksel + e] * sc);
                aq[s][8 + e] = (_Float16)(qp[s * 32 + aksel + 16 + e] * sc);
            }
    }

    v8f oacc[4];
    for (int dt = 0; dt < 4; ++dt)
        for (int j = 0; j < 8; ++j) oacc[dt][j] = 0.0f;
    float mrow[8], lrow[8];
    for (int j = 0; j < 8; ++j) { mrow[j] = -3.0e38f; lrow[j] = 0.0f; }

    _Float16* pw = &lP[wave * 16 * PSTR];
    const int kend = q0 + 128;

    for (int kb = 0; kb < kend; kb += 64) {
        __syncthreads();
        // Stage K block (64 x 64 fp32) via async DMA to LDS
        for (int it = 0; it < 4; ++it) {
            int i  = tid + it * 256;
            int kr = i >> 4;
            int c4 = (i & 15) * 4;
            async_copy_b128(kbase + (size_t)(kb + kr) * DIMSZ + c4,
                            &lK[kr * KSTRF + c4]);
        }
        // Stage V block transposed: 64 rows (k) x 64 cols (d) -> lVt[d][k], f16
        for (int it = 0; it < 4; ++it) {
            int i  = tid + it * 256;
            int kr = i >> 4;
            int c4 = (i & 15) * 4;
            float4 f = *(const float4*)(vbase + (size_t)(kb + kr) * DIMSZ + c4);
            lVt[(c4 + 0) * VSTR + kr] = (_Float16)f.x;
            lVt[(c4 + 1) * VSTR + kr] = (_Float16)f.y;
            lVt[(c4 + 2) * VSTR + kr] = (_Float16)f.z;
            lVt[(c4 + 3) * VSTR + kr] = (_Float16)f.w;
        }
        wait_async0();
        __syncthreads();

        // Scores: S = (Q*scale) @ K^T   (B-frag from LDS: contiguous d per k-row)
        v8f sacc[4];
        for (int nt = 0; nt < 4; ++nt) {
            for (int j = 0; j < 8; ++j) sacc[nt][j] = 0.0f;
            int krow = nt * 16 + l15;
            for (int s = 0; s < 2; ++s) {
                const float* kp = &lK[krow * KSTRF + bksel + s * 32];
                v16h bf;
                for (int e = 0; e < 16; ++e) bf[e] = (_Float16)kp[e < 8 ? e : e + 8];
                sacc[nt] = wmma_f16(aq[s], bf, sacc[nt]);
            }
        }
        // Causal mask
        for (int nt = 0; nt < 4; ++nt) {
            int col = kb + nt * 16 + l15;
            for (int j = 0; j < 8; ++j)
                if (col > qw + moff + j) sacc[nt][j] = -1.0e9f;
        }
        // Online softmax (row reductions across the 16 lanes of each half)
        float pj[4][8];
        for (int j = 0; j < 8; ++j) {
            float vmax = sacc[0][j];
            for (int nt = 1; nt < 4; ++nt) vmax = fmaxf(vmax, sacc[nt][j]);
            for (int off = 1; off < 16; off <<= 1)
                vmax = fmaxf(vmax, __shfl_xor(vmax, off, 32));
            float mnew  = fmaxf(mrow[j], vmax);
            float alpha = __expf(mrow[j] - mnew);
            float rsum  = 0.0f;
            for (int nt = 0; nt < 4; ++nt) {
                float p = __expf(sacc[nt][j] - mnew);
                pj[nt][j] = p;
                rsum += p;
            }
            for (int off = 1; off < 16; off <<= 1)
                rsum += __shfl_xor(rsum, off, 32);
            lrow[j] = lrow[j] * alpha + rsum;
            mrow[j] = mnew;
            for (int dt = 0; dt < 4; ++dt) oacc[dt][j] *= alpha;
        }
        // Re-layout P (C-layout -> A-layout) through per-wave LDS
        for (int nt = 0; nt < 4; ++nt)
            for (int j = 0; j < 8; ++j)
                pw[(moff + j) * PSTR + nt * 16 + l15] = (_Float16)pj[nt][j];
        asm volatile("s_wait_dscnt 0x0" ::: "memory");

        // O += P @ V
        for (int s = 0; s < 2; ++s) {
            v16h pf;
            const _Float16* pp = &pw[l15 * PSTR + aksel + s * 32];
            v8h plo = *(const v8h*)pp;
            v8h phi = *(const v8h*)(pp + 16);
            for (int e = 0; e < 8; ++e) { pf[e] = plo[e]; pf[8 + e] = phi[e]; }
            for (int dt = 0; dt < 4; ++dt) {
                const _Float16* vp = &lVt[(dt * 16 + l15) * VSTR + bksel + s * 32];
                v8h vlo = *(const v8h*)vp;
                v8h vhi = *(const v8h*)(vp + 16);
                v16h vf;
                for (int e = 0; e < 8; ++e) { vf[e] = vlo[e]; vf[8 + e] = vhi[e]; }
                oacc[dt] = wmma_f16(pf, vf, oacc[dt]);
            }
        }
    }

    for (int dt = 0; dt < 4; ++dt)
        for (int j = 0; j < 8; ++j) {
            int row = qw + moff + j;
            obase[(size_t)row * DIMSZ + dt * 16 + l15] = oacc[dt][j] / lrow[j];
        }
}

// ---------------------------------------------------------------------------
// out[row] = LayerNorm(a[row] + r[row]) * g + b   (one block per row of 1024)
// ---------------------------------------------------------------------------
__global__ __launch_bounds__(256) void add_ln_kernel(
    const float* __restrict__ a, const float* __restrict__ r,
    const float* __restrict__ g, const float* __restrict__ bb,
    float* __restrict__ out)
{
    __shared__ float red1[8];
    __shared__ float red2[8];
    const int row  = blockIdx.x;
    const int tid  = threadIdx.x;
    const int wave = tid >> 5;
    const int lane = tid & 31;
    const float* ap = a + (size_t)row * DIMSZ;
    const float* rp = r + (size_t)row * DIMSZ;

    float v[4];
    float s = 0.0f;
    for (int i = 0; i < 4; ++i) {
        int c = tid + i * 256;
        v[i] = ap[c] + rp[c];
        s += v[i];
    }
    for (int off = 16; off >= 1; off >>= 1) s += __shfl_xor(s, off, 32);
    if (lane == 0) red1[wave] = s;
    __syncthreads();
    float tot = 0.0f;
    for (int w = 0; w < 8; ++w) tot += red1[w];
    float mu = tot * (1.0f / DIMSZ);

    float sq = 0.0f;
    for (int i = 0; i < 4; ++i) { float d = v[i] - mu; sq += d * d; }
    for (int off = 16; off >= 1; off >>= 1) sq += __shfl_xor(sq, off, 32);
    if (lane == 0) red2[wave] = sq;
    __syncthreads();
    float tot2 = 0.0f;
    for (int w = 0; w < 8; ++w) tot2 += red2[w];
    float rstd = rsqrtf(tot2 * (1.0f / DIMSZ) + 1e-5f);

    for (int i = 0; i < 4; ++i) {
        int c = tid + i * 256;
        out[(size_t)row * DIMSZ + c] = (v[i] - mu) * rstd * g[c] + bb[c];
    }
}

// ---------------------------------------------------------------------------
extern "C" void kernel_launch(void* const* d_in, const int* in_sizes, int n_in,
                              void* d_out, int out_size, void* d_ws, size_t ws_size,
                              hipStream_t stream) {
    (void)in_sizes; (void)n_in; (void)out_size; (void)ws_size;
    const float* x     = (const float*)d_in[0];
    const float* Wq    = (const float*)d_in[1];
    const float* Wk    = (const float*)d_in[2];
    const float* Wv    = (const float*)d_in[3];
    const float* bq    = (const float*)d_in[4];
    const float* bk    = (const float*)d_in[5];
    const float* bv    = (const float*)d_in[6];
    const float* ln1_g = (const float*)d_in[7];
    const float* ln1_b = (const float*)d_in[8];
    const float* W1    = (const float*)d_in[9];
    const float* b1    = (const float*)d_in[10];
    const float* W2    = (const float*)d_in[11];
    const float* b2    = (const float*)d_in[12];
    const float* ln2_g = (const float*)d_in[13];
    const float* ln2_b = (const float*)d_in[14];

    float* ws = (float*)d_ws;
    const size_t SZ = (size_t)BATCH * SEQ * DIMSZ;        // 8M floats
    float* q    = ws + 0 * SZ;
    float* k    = ws + 1 * SZ;
    float* v    = ws + 2 * SZ;
    float* attn = ws + 3 * SZ;
    float* h    = ws + 4 * SZ;
    float* ffn2 = ws + 5 * SZ;
    float* ffn1 = ws;   // reuses q/k/v/attn region (4*SZ = 32M floats) after h exists

    const int MROWS = BATCH * SEQ;                        // 8192
    dim3 blk(256);

    dim3 gproj(DIMSZ / 128, MROWS / 128);
    gemm_wmma_kernel<<<gproj, blk, 0, stream>>>(x, Wq, bq, q, MROWS, DIMSZ, DIMSZ, 0);
    gemm_wmma_kernel<<<gproj, blk, 0, stream>>>(x, Wk, bk, k, MROWS, DIMSZ, DIMSZ, 0);
    gemm_wmma_kernel<<<gproj, blk, 0, stream>>>(x, Wv, bv, v, MROWS, DIMSZ, DIMSZ, 0);

    attn_kernel<<<dim3(SEQ / 128, NHEADS, BATCH), blk, 0, stream>>>(q, k, v, attn);

    add_ln_kernel<<<dim3(MROWS), blk, 0, stream>>>(attn, x, ln1_g, ln1_b, h);

    gemm_wmma_kernel<<<dim3(4 * DIMSZ / 128, MROWS / 128), blk, 0, stream>>>(
        h, W1, b1, ffn1, MROWS, 4 * DIMSZ, DIMSZ, 1);
    gemm_wmma_kernel<<<dim3(DIMSZ / 128, MROWS / 128), blk, 0, stream>>>(
        ffn1, W2, b2, ffn2, MROWS, DIMSZ, 4 * DIMSZ, 0);

    add_ln_kernel<<<dim3(MROWS), blk, 0, stream>>>(ffn2, h, ln2_g, ln2_b, (float*)d_out);
}
